// TT_MoE_Gate_23029614641842
// MI455X (gfx1250) — compile-verified, matrix operand order checked by reading, not run on hardware
//
#include <hip/hip_runtime.h>
#include <hip/hip_bf16.h>
#include <math.h>

#define N_TOK   8192
#define HDIM    7168
#define NEXP    256
#define SCALE_F 2.5f
#define EPS_F   1e-20f

typedef __attribute__((ext_vector_type(16))) __bf16 v16bf;
typedef __attribute__((ext_vector_type(8)))  float  v8f;
typedef __attribute__((ext_vector_type(4)))  float  v4f;   // clang vector: NT-load legal

// ---------------------------------------------------------------------------
// One-time: weight[H,E] f32 (E fast) -> wT[E,H] bf16 (H fast), LDS-tiled.
// wT rows become the WMMA B-fragment source: contiguous K per expert.
// ---------------------------------------------------------------------------
__global__ __launch_bounds__(256)
void TT_MoE_Gate_wconv(const float* __restrict__ w,
                       __hip_bfloat16* __restrict__ wt) {
  __shared__ __hip_bfloat16 tile[32][33];
  const int h0 = blockIdx.x * 32;
  const int e0 = blockIdx.y * 32;
  const int tx = threadIdx.x;   // 0..31
  const int ty = threadIdx.y;   // 0..7
  #pragma unroll
  for (int i = ty; i < 32; i += 8)
    tile[i][tx] = __float2bfloat16(w[(size_t)(h0 + i) * NEXP + (e0 + tx)]);
  __syncthreads();
  #pragma unroll
  for (int i = ty; i < 32; i += 8)
    wt[(size_t)(e0 + i) * HDIM + (h0 + tx)] = tile[tx][i];
}

// ---------------------------------------------------------------------------
// Fused gate: bf16 WMMA GEMM (32 tokens x 256 experts per block) -> LDS ->
// softmax + bias + group-limited top-k routing, all in one pass.
// Block = 128 threads = 4 waves; wave w owns experts [64w, 64w+64).
// ---------------------------------------------------------------------------
__global__ __launch_bounds__(128)
void TT_MoE_Gate_main(const float* __restrict__ x,
                      const __hip_bfloat16* __restrict__ wt,
                      const float* __restrict__ bias,
                      float* __restrict__ out) {
  __shared__ float slog[32 * 258];   // 33 KB; stride 258 avoids bank clashes

  const int tid   = threadIdx.x;
  const int wave  = tid >> 5;
  const int lane  = tid & 31;
  const int l16   = lane & 15;
  const int lhalf = lane >> 4;       // K-split half per WMMA layout
  const int tok0  = blockIdx.x * 32;
  const int ebase = wave * 64;

  // ---------------- GEMM phase: logits[32 x 256] ----------------
  v8f acc[2][4];
  #pragma unroll
  for (int tt = 0; tt < 2; ++tt)
    #pragma unroll
    for (int ct = 0; ct < 4; ++ct)
      #pragma unroll
      for (int r = 0; r < 8; ++r) acc[tt][ct][r] = 0.0f;

  const float* xr0 = x + (size_t)(tok0 + l16) * HDIM;          // A: M = l16
  const float* xr1 = xr0 + (size_t)16 * HDIM;
  const __hip_bfloat16* wr0 = wt + (size_t)(ebase +  0 + l16) * HDIM; // B: N = l16
  const __hip_bfloat16* wr1 = wt + (size_t)(ebase + 16 + l16) * HDIM;
  const __hip_bfloat16* wr2 = wt + (size_t)(ebase + 32 + l16) * HDIM;
  const __hip_bfloat16* wr3 = wt + (size_t)(ebase + 48 + l16) * HDIM;

  for (int k0 = 0; k0 < HDIM; k0 += 32) {
    // per-lane K base: lanes 0-15 take K {0..7,16..23}, lanes 16-31 {8..15,24..31}
    const int kb = k0 + (lhalf << 3);

    // Unconditional stream prefetch (no branch -> loop stays one basic block
    // so the scheduler can software-pipeline next-iteration loads). Each lane
    // advances 128 B/iter, so this touches a fresh line 8 KB ahead each time;
    // speculative prefetch past the row tail is silently dropped.
    __builtin_prefetch(xr0 + kb + 2048, 0, 0);
    __builtin_prefetch(xr1 + kb + 2048, 0, 0);

    // A fragments: non-temporal fp32 loads (x is read exactly once ->
    // don't churn L2, which must keep the 3.7MB wT resident), cvt to bf16
    v16bf afrag[2];
    #pragma unroll
    for (int tt = 0; tt < 2; ++tt) {
      const float* p = (tt ? xr1 : xr0) + kb;
      const v4f f0 = __builtin_nontemporal_load((const v4f*)(p +  0));
      const v4f f1 = __builtin_nontemporal_load((const v4f*)(p +  4));
      const v4f f2 = __builtin_nontemporal_load((const v4f*)(p + 16));
      const v4f f3 = __builtin_nontemporal_load((const v4f*)(p + 20));
      v16bf a;
      a[ 0] = (__bf16)f0[0];  a[ 1] = (__bf16)f0[1];
      a[ 2] = (__bf16)f0[2];  a[ 3] = (__bf16)f0[3];
      a[ 4] = (__bf16)f1[0];  a[ 5] = (__bf16)f1[1];
      a[ 6] = (__bf16)f1[2];  a[ 7] = (__bf16)f1[3];
      a[ 8] = (__bf16)f2[0];  a[ 9] = (__bf16)f2[1];
      a[10] = (__bf16)f2[2];  a[11] = (__bf16)f2[3];
      a[12] = (__bf16)f3[0];  a[13] = (__bf16)f3[1];
      a[14] = (__bf16)f3[2];  a[15] = (__bf16)f3[3];
      afrag[tt] = a;
    }

    const __hip_bfloat16* wr[4] = {wr0, wr1, wr2, wr3};
    #pragma unroll
    for (int ct = 0; ct < 4; ++ct) {
      union { v16bf v; uint4 q[2]; } bu;
      bu.q[0] = *(const uint4*)(wr[ct] + kb);       // bf16 already: straight b128
      bu.q[1] = *(const uint4*)(wr[ct] + kb + 16);
      acc[0][ct] = __builtin_amdgcn_wmma_f32_16x16x32_bf16(
          false, afrag[0], false, bu.v, (short)0, acc[0][ct], false, false);
      acc[1][ct] = __builtin_amdgcn_wmma_f32_16x16x32_bf16(
          false, afrag[1], false, bu.v, (short)0, acc[1][ct], false, false);
    }
  }

  // C layout: VGPR r, lane l -> M = r + 8*(l>=16), N = l%16
  #pragma unroll
  for (int tt = 0; tt < 2; ++tt)
    #pragma unroll
    for (int ct = 0; ct < 4; ++ct)
      #pragma unroll
      for (int r = 0; r < 8; ++r) {
        const int m = tt * 16 + lhalf * 8 + r;
        const int n = ebase + ct * 16 + l16;
        slog[m * 258 + n] = acc[tt][ct][r];
      }
  __syncthreads();

  // ---------------- Routing phase: one wave per token, 8 experts/lane -------
  float bl[8];
  #pragma unroll
  for (int j = 0; j < 8; ++j) bl[j] = bias[lane * 8 + j];

  for (int t = wave; t < 32; t += 4) {
    const float* row = slog + t * 258;
    float s[8];
    #pragma unroll
    for (int j = 0; j < 8; ++j) s[j] = row[lane * 8 + j];

    // softmax over 256
    float mx = s[0];
    #pragma unroll
    for (int j = 1; j < 8; ++j) mx = fmaxf(mx, s[j]);
    #pragma unroll
    for (int d = 16; d > 0; d >>= 1) mx = fmaxf(mx, __shfl_xor(mx, d, 32));
    float p[8], sum = 0.0f;
    #pragma unroll
    for (int j = 0; j < 8; ++j) { p[j] = __expf(s[j] - mx); sum += p[j]; }
    #pragma unroll
    for (int d = 16; d > 0; d >>= 1) sum += __shfl_xor(sum, d, 32);
    const float inv = 1.0f / sum;

    float score[8], sb[8];
    #pragma unroll
    for (int j = 0; j < 8; ++j) { score[j] = p[j] * inv; sb[j] = score[j] + bl[j]; }

    // group score = sum of top-2 biased scores; group g = lane quad g
    float a1 = -INFINITY, a2 = -INFINITY;
    #pragma unroll
    for (int j = 0; j < 8; ++j) {
      const float v  = sb[j];
      const float hi = fmaxf(a1, v);
      a2 = fmaxf(a2, fminf(a1, v));
      a1 = hi;
    }
    #pragma unroll
    for (int d = 1; d <= 2; d <<= 1) {   // merge top-2 across the quad
      const float b1 = __shfl_xor(a1, d, 32);
      const float b2 = __shfl_xor(a2, d, 32);
      const float hi = fmaxf(a1, b1);
      const float lo = fmaxf(fminf(a1, b1), fmaxf(a2, b2));
      a1 = hi; a2 = lo;
    }
    const float gs = a1 + a2;
    const int   q  = lane >> 2;
    int rank = 0;
    #pragma unroll
    for (int g = 0; g < 8; ++g) {        // stable rank -> top-4 groups
      const float gsg = __shfl(gs, g * 4, 32);
      rank += (gsg > gs) || ((gsg == gs) && (g < q));
    }
    const bool keep = rank < 4;

    float masked[8];
    #pragma unroll
    for (int j = 0; j < 8; ++j) masked[j] = keep ? sb[j] : -INFINITY;

    // stable top-8 experts: 8 rounds of wave-wide argmax carrying unbiased score
    float wsel[8]; int isel[8]; float wsum = 0.0f;
    #pragma unroll
    for (int r = 0; r < 8; ++r) {
      float v = -INFINITY; int idx = 0x7FFFFFFF; float sc = 0.0f;
      #pragma unroll
      for (int j = 0; j < 8; ++j)
        if (masked[j] > v) { v = masked[j]; idx = lane * 8 + j; sc = score[j]; }
      #pragma unroll
      for (int d = 16; d > 0; d >>= 1) {
        const float ov = __shfl_xor(v, d, 32);
        const int   oi = __shfl_xor(idx, d, 32);
        const float os = __shfl_xor(sc, d, 32);
        if (ov > v || (ov == v && oi < idx)) { v = ov; idx = oi; sc = os; }
      }
      wsel[r] = sc; isel[r] = idx; wsum += sc;
      if ((idx >> 3) == lane) {          // owner retires the winner
        const int jj = idx & 7;
        #pragma unroll
        for (int j = 0; j < 8; ++j) if (j == jj) masked[j] = -INFINITY;
      }
    }

    if (lane == 0) {
      const float rs  = 1.0f / wsum;
      const int   tok = tok0 + t;
      #pragma unroll
      for (int r = 0; r < 8; ++r) {
        out[(size_t)tok * 8 + r] = (wsel[r] * rs + EPS_F) * SCALE_F;          // w
        out[(size_t)N_TOK * 8 + (size_t)tok * 8 + r] = (float)isel[r];        // idx
      }
    }
  }
}

extern "C" void kernel_launch(void* const* d_in, const int* in_sizes, int n_in,
                              void* d_out, int out_size, void* d_ws, size_t ws_size,
                              hipStream_t stream) {
  (void)in_sizes; (void)n_in; (void)out_size; (void)ws_size;
  const float* x    = (const float*)d_in[0];
  const float* w    = (const float*)d_in[1];
  const float* bias = (const float*)d_in[2];
  float* out = (float*)d_out;
  __hip_bfloat16* wt = (__hip_bfloat16*)d_ws;   // needs 256*7168*2 = 3.67 MB scratch

  dim3 tg(HDIM / 32, NEXP / 32), tb(32, 8);
  TT_MoE_Gate_wconv<<<tg, tb, 0, stream>>>(w, wt);
  TT_MoE_Gate_main<<<N_TOK / 32, 128, 0, stream>>>(x, wt, bias, out);
}
